// EmbedDTI_Pre_36850819400351
// MI455X (gfx1250) — matrix operand order repaired and checked by reading, not compiled, not run on hardware
//
#include <hip/hip_runtime.h>
#include <hip/hip_bf16.h>

typedef __attribute__((ext_vector_type(16))) _Float16 v16h;
typedef __attribute__((ext_vector_type(8)))  float    v8f;

#define BM 128
#define BN 64
#define BK 32

// ---------------------------------------------------------------------------
// LDS fragment swizzle helpers (match documented 16-bit WMMA VGPR layouts).
// A tile element (r, c) -> fragment (r>>4), lane (r&15)+16h, elem e
//   where h = (c>>3)&1 and e maps {0..7,16..23} -> {0..15}.
// B tile element (k, n) -> fragment (n>>4), lane (n&15)+16*((k>>3)&1), elem (k&7)+8*(k>>4).
// Each lane's 16 fp16 elements are contiguous (32B) => 2x ds_load_b128.
// stash_a2 writes an even-aligned (c, c+1) pair as one 32-bit LDS store.
// ---------------------------------------------------------------------------
__device__ __forceinline__ void stash_a2(v16h* As_f, int r, int c, float x0, float x1) {
  int h  = (c >> 3) & 1;
  int c2 = c - (h << 3);
  int e  = (c2 < 8) ? c2 : c2 - 8;              // even
  int frag = ((r >> 4) << 5) + (r & 15) + (h << 4);
  union { _Float16 h2[2]; unsigned u; } pk;
  pk.h2[0] = (_Float16)x0; pk.h2[1] = (_Float16)x1;
  *(unsigned*)((_Float16*)&As_f[frag] + e) = pk.u;
}
__device__ __forceinline__ void stash_b(v16h* Bs_f, int k, int n, float x) {
  int frag = ((n >> 4) << 5) + (n & 15) + (((k >> 3) & 1) << 4);
  ((_Float16*)&Bs_f[frag])[(k & 7) + ((k >> 4) << 3)] = (_Float16)x;
}

// ---------------------------------------------------------------------------
// Generic WMMA GEMM: C[m, c_off + n] = act(A[M,K] @ B[K,N] + bias[n])
// A, B fp32 row-major; staged to LDS as fp16 in fragment order.
// Interior tiles take a uniform guard-free fast path; edge tiles use clamped
// addresses with arithmetic masking so loads stay unconditional.
// Block: 256 threads = 8 waves; tile 128x64; each wave owns 16 rows x 64 cols.
// ---------------------------------------------------------------------------
__global__ __launch_bounds__(256) void gemm_wmma(
    const float* __restrict__ A, const float* __restrict__ B,
    const float* __restrict__ bias, float* __restrict__ C,
    int M, int N, int K, int ldc, int c_off, int relu)
{
  __shared__ v16h As_f[256];   // 8 row-groups x 32 lanes, 8 KB
  __shared__ v16h Bs_f[128];   // 4 col-groups x 32 lanes, 4 KB

  const int tid  = threadIdx.x;
  const int wave = tid >> 5;
  const int lane = tid & 31;
  const int m0 = blockIdx.y * BM;
  const int n0 = blockIdx.x * BN;
  const bool rows_full = (m0 + BM) <= M;
  const bool cols_full = (n0 + BN) <= N;

  v8f acc[4];
  #pragma unroll
  for (int j = 0; j < 4; ++j) acc[j] = (v8f){0,0,0,0,0,0,0,0};

  for (int k0 = 0; k0 < K; k0 += BK) {
    const bool k_full = (k0 + BK) <= K;
    // ---- A tile (128x32): 8 pairs/thread along contiguous K ----
    if (rows_full && k_full) {
      #pragma unroll
      for (int it = 0; it < (BM * BK) / 512; ++it) {
        int p = tid + it * 256;
        int r = p >> 4, c = (p & 15) << 1;
        const float* Ap = &A[(long)(m0 + r) * K + (k0 + c)];
        stash_a2(As_f, r, c, Ap[0], Ap[1]);
      }
    } else {
      #pragma unroll
      for (int it = 0; it < (BM * BK) / 512; ++it) {
        int p = tid + it * 256;
        int r = p >> 4, c = (p & 15) << 1;
        int gm = m0 + r, gk = k0 + c;
        int cm  = (gm < M) ? gm : (M - 1);
        int ck0 = (gk < K) ? gk : (K - 1);
        int ck1 = (gk + 1 < K) ? (gk + 1) : (K - 1);
        float f0 = (gm < M && gk < K) ? 1.0f : 0.0f;
        float f1 = (gm < M && gk + 1 < K) ? 1.0f : 0.0f;
        float x0 = A[(long)cm * K + ck0] * f0;
        float x1 = A[(long)cm * K + ck1] * f1;
        stash_a2(As_f, r, c, x0, x1);
      }
    }
    // ---- B tile (32x64): 4 pairs/thread along contiguous N ----
    if (cols_full && k_full) {
      #pragma unroll
      for (int it = 0; it < (BK * BN) / 512; ++it) {
        int p = tid + it * 256;
        int k = p >> 5, c = (p & 31) << 1;
        const float* Bp = &B[(long)(k0 + k) * N + (n0 + c)];
        stash_b(Bs_f, k, c, Bp[0]);
        stash_b(Bs_f, k, c + 1, Bp[1]);
      }
    } else {
      #pragma unroll
      for (int it = 0; it < (BK * BN) / 512; ++it) {
        int p = tid + it * 256;
        int k = p >> 5, c = (p & 31) << 1;
        int gk = k0 + k, gn = n0 + c;
        int ck  = (gk < K) ? gk : (K - 1);
        int cn0 = (gn < N) ? gn : (N - 1);
        int cn1 = (gn + 1 < N) ? (gn + 1) : (N - 1);
        float f0 = (gk < K && gn < N) ? 1.0f : 0.0f;
        float f1 = (gk < K && gn + 1 < N) ? 1.0f : 0.0f;
        stash_b(Bs_f, k, c, B[(long)ck * N + cn0] * f0);
        stash_b(Bs_f, k, c + 1, B[(long)ck * N + cn1] * f1);
      }
    }
    if (k0 + BK < K) {
      __builtin_prefetch(&A[(long)(m0 + (tid & 127)) * K + (k0 + BK)], 0, 1);
      __builtin_prefetch(&B[(long)(k0 + BK + (tid & 31)) * N + n0], 0, 1);
    }
    __syncthreads();

    v16h a = As_f[(wave << 5) + lane];      // 2x ds_load_b128
    #pragma unroll
    for (int j = 0; j < 4; ++j) {
      v16h b = Bs_f[(j << 5) + lane];       // 2x ds_load_b128
      acc[j] = __builtin_amdgcn_wmma_f32_16x16x32_f16(
          false, a, false, b, (short)0, acc[j], false, false);
    }
    __syncthreads();
  }

  const int half = lane >> 4, lr = lane & 15;
  #pragma unroll
  for (int j = 0; j < 4; ++j) {
    #pragma unroll
    for (int q = 0; q < 8; ++q) {
      int gm = m0 + wave * 16 + q + 8 * half;   // C layout: VGPR q -> row q / q+8
      int gn = n0 + j * 16 + lr;
      if (gm < M && gn < N) {
        float val = acc[j][q] + (bias ? bias[gn] : 0.0f);
        if (relu) val = fmaxf(val, 0.0f);
        C[(long)gm * ldc + c_off + gn] = val;
      }
    }
  }
}

// ---------------------------------------------------------------------------
// WMMA conv1d-as-GEMM (VALID, stride 1). Per batch item b (grid.z):
//   Out[b, co, l] = sum_{ci,k} W[co, ci*Kw+k] * In[b, ci, l+k]
// im2col column (j = ci*Kw+k) is contiguous in l, so B tiles are gathered
// directly. use_embed: In[b,ci,:] := emb[target[b*C_in+ci], :]  (L_in = 128).
// K = C_in*Kw is a multiple of BK for all three convs (8000, 2048, 96).
// ---------------------------------------------------------------------------
__global__ __launch_bounds__(256) void conv_wmma(
    const float* __restrict__ W, const float* __restrict__ In,
    const int* __restrict__ target, const float* __restrict__ emb,
    const float* __restrict__ bias, float* __restrict__ Out,
    int C_out, int C_in, int Kw, int L_in, int L_out, int use_embed)
{
  __shared__ v16h As_f[256];
  __shared__ v16h Bs_f[128];

  const int tid  = threadIdx.x;
  const int wave = tid >> 5;
  const int lane = tid & 31;
  const int m0 = blockIdx.y * BM;       // C_out tile
  const int n0 = blockIdx.x * BN;       // L_out tile
  const int b  = blockIdx.z;
  const int K  = C_in * Kw;
  const bool rows_full = (m0 + BM) <= C_out;
  const bool cols_full = (n0 + BN) <= L_out;

  v8f acc[4];
  #pragma unroll
  for (int j = 0; j < 4; ++j) acc[j] = (v8f){0,0,0,0,0,0,0,0};

  for (int k0 = 0; k0 < K; k0 += BK) {
    // ---- A tile: weights [C_out, K], K % BK == 0 so only row guard ----
    if (rows_full) {
      #pragma unroll
      for (int it = 0; it < (BM * BK) / 512; ++it) {
        int p = tid + it * 256;
        int r = p >> 4, c = (p & 15) << 1;
        const float* Wp = &W[(long)(m0 + r) * K + (k0 + c)];
        stash_a2(As_f, r, c, Wp[0], Wp[1]);
      }
    } else {
      #pragma unroll
      for (int it = 0; it < (BM * BK) / 512; ++it) {
        int p = tid + it * 256;
        int r = p >> 4, c = (p & 15) << 1;
        int gm = m0 + r;
        int cm = (gm < C_out) ? gm : (C_out - 1);
        float f = (gm < C_out) ? 1.0f : 0.0f;
        const float* Wp = &W[(long)cm * K + (k0 + c)];
        stash_a2(As_f, r, c, Wp[0] * f, Wp[1] * f);
      }
    }
    // ---- B tile: shifted-gather, contiguous along l ----
    #pragma unroll
    for (int it = 0; it < (BK * BN) / 512; ++it) {
      int p = tid + it * 256;
      int k = p >> 5, c = (p & 31) << 1;
      int gk = k0 + k, gn = n0 + c;
      int ci = gk / Kw, kk = gk - ci * Kw;
      const float* row;
      if (use_embed) {
        int t = target[(long)b * C_in + ci];
        row = &emb[(long)t * L_in + kk];
      } else {
        row = &In[((long)b * C_in + ci) * (long)L_in + kk];
      }
      if (cols_full) {
        stash_b(Bs_f, k, c, row[gn]);
        stash_b(Bs_f, k, c + 1, row[gn + 1]);
      } else {
        int ln0 = (gn < L_out) ? gn : 0;
        int ln1 = (gn + 1 < L_out) ? (gn + 1) : 0;
        float f0 = (gn < L_out) ? 1.0f : 0.0f;
        float f1 = (gn + 1 < L_out) ? 1.0f : 0.0f;
        stash_b(Bs_f, k, c, row[ln0] * f0);
        stash_b(Bs_f, k, c + 1, row[ln1] * f1);
      }
    }
    __syncthreads();

    v16h a = As_f[(wave << 5) + lane];
    #pragma unroll
    for (int j = 0; j < 4; ++j) {
      v16h bb = Bs_f[(j << 5) + lane];
      acc[j] = __builtin_amdgcn_wmma_f32_16x16x32_f16(
          false, a, false, bb, (short)0, acc[j], false, false);
    }
    __syncthreads();
  }

  const int half = lane >> 4, lr = lane & 15;
  #pragma unroll
  for (int j = 0; j < 4; ++j) {
    #pragma unroll
    for (int q = 0; q < 8; ++q) {
      int gm = m0 + wave * 16 + q + 8 * half;
      int gn = n0 + j * 16 + lr;
      if (gm < C_out && gn < L_out) {
        Out[((long)b * C_out + gm) * (long)L_out + gn] = acc[j][q] + bias[gm];
      }
    }
  }
}

// ---------------------------------------------------------------------------
// GCN aggregation helpers (bandwidth/atomic bound; plain VMEM path)
// ---------------------------------------------------------------------------
__global__ void fill_f32(float* __restrict__ p, float v, long n) {
  long i = (long)blockIdx.x * blockDim.x + threadIdx.x;
  if (i < n) p[i] = v;
}

__global__ void deg_accum(const int* __restrict__ dst, float* __restrict__ deg, int E) {
  int e = blockIdx.x * blockDim.x + threadIdx.x;
  if (e < E) atomicAdd(&deg[dst[e]], 1.0f);
}

__global__ void rsqrt_inplace(float* __restrict__ deg, int n) {
  int i = blockIdx.x * blockDim.x + threadIdx.x;
  if (i < n) deg[i] = rsqrtf(fmaxf(deg[i], 1.0f));   // stores dis = deg^-1/2
}

// agg[d,:] += h[s,:] * dis[s]*dis[d]   (one block per edge)
__global__ void edge_scatter(const int* __restrict__ src, const int* __restrict__ dst,
                             const float* __restrict__ dis, const float* __restrict__ h,
                             float* __restrict__ agg, int F, int E) {
  int e = blockIdx.x;
  if (e >= E) return;
  int s = src[e], d = dst[e];
  float nrm = dis[s] * dis[d];
  const float* hs = h + (long)s * F;
  float* ad = agg + (long)d * F;
  for (int f = threadIdx.x; f < F; f += blockDim.x)
    atomicAdd(&ad[f], hs[f] * nrm);
}

// out = relu(agg + hW * dis^2 (self loop) + bias), in place on agg
__global__ void gcn_finalize(const float* __restrict__ hW, const float* __restrict__ dis,
                             const float* __restrict__ bias, float* __restrict__ agg,
                             int F, long Nn) {
  long i = (long)blockIdx.x * blockDim.x + threadIdx.x;
  if (i < Nn * F) {
    long node = i / F; int f = (int)(i - node * F);
    float dn = dis[node];
    float v = agg[i] + hW[i] * dn * dn + bias[f];
    agg[i] = fmaxf(v, 0.0f);
  }
}

// segment max via int-bit atomicMax (valid: inputs are post-ReLU, >= 0)
__global__ void seg_max(const float* __restrict__ h, const int* __restrict__ batch,
                        float* __restrict__ pool, int F, long Nn) {
  long i = (long)blockIdx.x * blockDim.x + threadIdx.x;
  if (i < Nn * F) {
    long node = i / F; int f = (int)(i - node * F);
    int b = batch[node];
    atomicMax((int*)&pool[(long)b * F + f], __float_as_int(h[i]));
  }
}

// out[b] = t2[b,:] . oW + ob   (K = 512)
__global__ __launch_bounds__(256) void final_dot(const float* __restrict__ t2,
                                                 const float* __restrict__ oW,
                                                 const float* __restrict__ ob,
                                                 float* __restrict__ out) {
  __shared__ float red[256];
  int b = blockIdx.x;
  float s = 0.0f;
  for (int k = threadIdx.x; k < 512; k += 256) s += t2[(long)b * 512 + k] * oW[k];
  red[threadIdx.x] = s;
  __syncthreads();
  for (int off = 128; off > 0; off >>= 1) {
    if (threadIdx.x < off) red[threadIdx.x] += red[threadIdx.x + off];
    __syncthreads();
  }
  if (threadIdx.x == 0) out[b] = red[0] + ob[0];
}

// ---------------------------------------------------------------------------
// Orchestration
// ---------------------------------------------------------------------------
#define N_ATOM 65536
#define E_ATOM 262144
#define N_CLQ  32768
#define E_CLQ  131072
#define BATCH  512

static inline dim3 gemm_grid(int M, int N) {
  return dim3((N + BN - 1) / BN, (M + BM - 1) / BM, 1);
}

extern "C" void kernel_launch(void* const* d_in, const int* in_sizes, int n_in,
                              void* d_out, int out_size, void* d_ws, size_t ws_size,
                              hipStream_t stream) {
  (void)in_sizes; (void)n_in; (void)out_size; (void)ws_size;

  const float* x        = (const float*)d_in[0];
  const int*   ei       = (const int*)  d_in[1];
  const int*   batch    = (const int*)  d_in[2];
  const float* clq_x    = (const float*)d_in[3];
  const int*   cei      = (const int*)  d_in[4];
  const int*   cbatch   = (const int*)  d_in[5];
  const int*   target   = (const int*)  d_in[6];
  const float* dW1 = (const float*)d_in[7];  const float* db1 = (const float*)d_in[8];
  const float* dW2 = (const float*)d_in[9];  const float* db2 = (const float*)d_in[10];
  const float* dW3 = (const float*)d_in[11]; const float* db3 = (const float*)d_in[12];
  const float* fcg1W = (const float*)d_in[13]; const float* fcg1b = (const float*)d_in[14];
  const float* fcg2W = (const float*)d_in[15]; const float* fcg2b = (const float*)d_in[16];
  const float* cW1 = (const float*)d_in[17]; const float* cb1 = (const float*)d_in[18];
  const float* cW2 = (const float*)d_in[19]; const float* cb2 = (const float*)d_in[20];
  const float* cW3 = (const float*)d_in[21]; const float* cb3 = (const float*)d_in[22];
  const float* cg1W = (const float*)d_in[23]; const float* cg1b = (const float*)d_in[24];
  const float* cg2W = (const float*)d_in[25]; const float* cg2b = (const float*)d_in[26];
  const float* emb  = (const float*)d_in[27];
  const float* cx1W = (const float*)d_in[28]; const float* cx1b = (const float*)d_in[29];
  const float* cx2W = (const float*)d_in[30]; const float* cx2b = (const float*)d_in[31];
  const float* cx3W = (const float*)d_in[32]; const float* cx3b = (const float*)d_in[33];
  const float* fxtW = (const float*)d_in[34]; const float* fxtb = (const float*)d_in[35];
  const float* f1W  = (const float*)d_in[36]; const float* f1b  = (const float*)d_in[37];
  const float* f2W  = (const float*)d_in[38]; const float* f2b  = (const float*)d_in[39];
  const float* oW   = (const float*)d_in[40]; const float* ob   = (const float*)d_in[41];

  const int* src  = ei;             const int* dst  = ei + E_ATOM;
  const int* csrc = cei;            const int* cdst = cei + E_CLQ;

  // ---- workspace layout (floats) ----
  const long BIGSZ = (long)N_ATOM * 404;           // 26,476,544
  float* W0 = (float*)d_ws;
  float* B0 = W0;
  float* B1 = W0 + BIGSZ;
  float* B2 = W0 + 2 * BIGSZ;
  float* deg_a = W0 + 3 * BIGSZ;
  float* deg_c = deg_a + N_ATOM;
  float* g_pool = deg_c + N_CLQ;                   // [512, 404]
  float* q_pool = g_pool + (long)BATCH * 404;      // [512, 140]
  float* gt1    = q_pool + (long)BATCH * 140;      // [512, 1024]
  float* qt1    = gt1 + (long)BATCH * 1024;        // [512, 1024]
  float* zc     = qt1 + (long)BATCH * 1024;        // [512, 384]
  float* t1     = zc  + (long)BATCH * 384;         // [512, 1024]
  float* t2     = t1  + (long)BATCH * 1024;        // [512, 512]
  float* out    = (float*)d_out;                   // [512, 1]

  const dim3 thr(256);
  auto cld = [](long n, int b) { return (unsigned)((n + b - 1) / b); };

  // ================= atom GCN branch =================
  fill_f32<<<cld(N_ATOM, 256), thr, 0, stream>>>(deg_a, 1.0f, N_ATOM);
  deg_accum<<<cld(E_ATOM, 256), thr, 0, stream>>>(dst, deg_a, E_ATOM);
  rsqrt_inplace<<<cld(N_ATOM, 256), thr, 0, stream>>>(deg_a, N_ATOM);

  // L1: 101 -> 101
  gemm_wmma<<<gemm_grid(N_ATOM, 101), thr, 0, stream>>>(x, dW1, nullptr, B0,
      N_ATOM, 101, 101, 101, 0, 0);
  fill_f32<<<cld((long)N_ATOM * 101, 256), thr, 0, stream>>>(B1, 0.0f, (long)N_ATOM * 101);
  edge_scatter<<<E_ATOM, dim3(128), 0, stream>>>(src, dst, deg_a, B0, B1, 101, E_ATOM);
  gcn_finalize<<<cld((long)N_ATOM * 101, 256), thr, 0, stream>>>(B0, deg_a, db1, B1, 101, N_ATOM);
  // L2: 101 -> 202
  gemm_wmma<<<gemm_grid(N_ATOM, 202), thr, 0, stream>>>(B1, dW2, nullptr, B0,
      N_ATOM, 202, 101, 202, 0, 0);
  fill_f32<<<cld((long)N_ATOM * 202, 256), thr, 0, stream>>>(B2, 0.0f, (long)N_ATOM * 202);
  edge_scatter<<<E_ATOM, dim3(128), 0, stream>>>(src, dst, deg_a, B0, B2, 202, E_ATOM);
  gcn_finalize<<<cld((long)N_ATOM * 202, 256), thr, 0, stream>>>(B0, deg_a, db2, B2, 202, N_ATOM);
  // L3: 202 -> 404
  gemm_wmma<<<gemm_grid(N_ATOM, 404), thr, 0, stream>>>(B2, dW3, nullptr, B0,
      N_ATOM, 404, 202, 404, 0, 0);
  fill_f32<<<cld((long)N_ATOM * 404, 256), thr, 0, stream>>>(B1, 0.0f, (long)N_ATOM * 404);
  edge_scatter<<<E_ATOM, dim3(128), 0, stream>>>(src, dst, deg_a, B0, B1, 404, E_ATOM);
  gcn_finalize<<<cld((long)N_ATOM * 404, 256), thr, 0, stream>>>(B0, deg_a, db3, B1, 404, N_ATOM);
  // global max pool
  fill_f32<<<cld((long)BATCH * 404, 256), thr, 0, stream>>>(g_pool, 0.0f, (long)BATCH * 404);
  seg_max<<<cld((long)N_ATOM * 404, 256), thr, 0, stream>>>(B1, batch, g_pool, 404, N_ATOM);

  // ================= clique GCN branch =================
  fill_f32<<<cld(N_CLQ, 256), thr, 0, stream>>>(deg_c, 1.0f, N_CLQ);
  deg_accum<<<cld(E_CLQ, 256), thr, 0, stream>>>(cdst, deg_c, E_CLQ);
  rsqrt_inplace<<<cld(N_CLQ, 256), thr, 0, stream>>>(deg_c, N_CLQ);

  gemm_wmma<<<gemm_grid(N_CLQ, 35), thr, 0, stream>>>(clq_x, cW1, nullptr, B0,
      N_CLQ, 35, 35, 35, 0, 0);
  fill_f32<<<cld((long)N_CLQ * 35, 256), thr, 0, stream>>>(B1, 0.0f, (long)N_CLQ * 35);
  edge_scatter<<<E_CLQ, dim3(128), 0, stream>>>(csrc, cdst, deg_c, B0, B1, 35, E_CLQ);
  gcn_finalize<<<cld((long)N_CLQ * 35, 256), thr, 0, stream>>>(B0, deg_c, cb1, B1, 35, N_CLQ);

  gemm_wmma<<<gemm_grid(N_CLQ, 70), thr, 0, stream>>>(B1, cW2, nullptr, B0,
      N_CLQ, 70, 35, 70, 0, 0);
  fill_f32<<<cld((long)N_CLQ * 70, 256), thr, 0, stream>>>(B2, 0.0f, (long)N_CLQ * 70);
  edge_scatter<<<E_CLQ, dim3(128), 0, stream>>>(csrc, cdst, deg_c, B0, B2, 70, E_CLQ);
  gcn_finalize<<<cld((long)N_CLQ * 70, 256), thr, 0, stream>>>(B0, deg_c, cb2, B2, 70, N_CLQ);

  gemm_wmma<<<gemm_grid(N_CLQ, 140), thr, 0, stream>>>(B2, cW3, nullptr, B0,
      N_CLQ, 140, 70, 140, 0, 0);
  fill_f32<<<cld((long)N_CLQ * 140, 256), thr, 0, stream>>>(B1, 0.0f, (long)N_CLQ * 140);
  edge_scatter<<<E_CLQ, dim3(128), 0, stream>>>(csrc, cdst, deg_c, B0, B1, 140, E_CLQ);
  gcn_finalize<<<cld((long)N_CLQ * 140, 256), thr, 0, stream>>>(B0, deg_c, cb3, B1, 140, N_CLQ);

  fill_f32<<<cld((long)BATCH * 140, 256), thr, 0, stream>>>(q_pool, 0.0f, (long)BATCH * 140);
  seg_max<<<cld((long)N_CLQ * 140, 256), thr, 0, stream>>>(B1, cbatch, q_pool, 140, N_CLQ);

  // ================= protein CNN branch (WMMA convs) =================
  // conv1: [512,1000,128(emb)] -> [512,256,121], K = 8000, gathered from emb table
  conv_wmma<<<dim3(2, 2, BATCH), thr, 0, stream>>>(cx1W, nullptr, target, emb, cx1b, B0,
      256, 1000, 8, 128, 121, 1);
  // conv2: [512,256,121] -> [512,32,114], K = 2048
  conv_wmma<<<dim3(2, 1, BATCH), thr, 0, stream>>>(cx2W, B0, nullptr, nullptr, cx2b, B1,
      32, 256, 8, 121, 114, 0);
  // conv3: [512,32,114] -> [512,32,112], K = 96
  conv_wmma<<<dim3(2, 1, BATCH), thr, 0, stream>>>(cx3W, B1, nullptr, nullptr, cx3b, B2,
      32, 32, 3, 114, 112, 0);

  // ================= FC heads into concatenated zc [512, 384] =================
  gemm_wmma<<<gemm_grid(BATCH, 1024), thr, 0, stream>>>(g_pool, fcg1W, fcg1b, gt1,
      BATCH, 1024, 404, 1024, 0, 1);
  gemm_wmma<<<gemm_grid(BATCH, 128), thr, 0, stream>>>(gt1, fcg2W, fcg2b, zc,
      BATCH, 128, 1024, 384, 0, 0);                 // cols 0..127
  gemm_wmma<<<gemm_grid(BATCH, 128), thr, 0, stream>>>(B2, fxtW, fxtb, zc,
      BATCH, 128, 3584, 384, 128, 0);               // cols 128..255 (c3 flattened)
  gemm_wmma<<<gemm_grid(BATCH, 1024), thr, 0, stream>>>(q_pool, cg1W, cg1b, qt1,
      BATCH, 1024, 140, 1024, 0, 1);
  gemm_wmma<<<gemm_grid(BATCH, 128), thr, 0, stream>>>(qt1, cg2W, cg2b, zc,
      BATCH, 128, 1024, 384, 256, 0);               // cols 256..383

  // ================= fused head =================
  gemm_wmma<<<gemm_grid(BATCH, 1024), thr, 0, stream>>>(zc, f1W, f1b, t1,
      BATCH, 1024, 384, 1024, 0, 1);
  gemm_wmma<<<gemm_grid(BATCH, 512), thr, 0, stream>>>(t1, f2W, f2b, t2,
      BATCH, 512, 1024, 512, 0, 1);
  final_dot<<<BATCH, thr, 0, stream>>>(t2, oW, ob, out);
}
